// DiqqetBashi_13950053778139
// MI455X (gfx1250) — matrix operand order repaired and checked by reading, not compiled, not run on hardware
//
#include <hip/hip_runtime.h>
#include <hip/hip_bf16.h>

// Fused single-head causal attention for MI455X (gfx1250, wave32, WMMA).
// B=4096, T=64, C=128, H=64.  One workgroup per batch, 8 waves (256 thr).
// All GEMMs via v_wmma_f32_16x16x32_f16 (fp32 accumulate), softmax in fp32.

typedef __attribute__((ext_vector_type(16))) _Float16 v16h;
typedef __attribute__((ext_vector_type(8)))  float    v8f;
typedef __attribute__((ext_vector_type(4)))  float    v4f;
typedef __attribute__((ext_vector_type(4)))  _Float16 v4h;

#define NB 4096
#define NT 64
#define NC 128
#define NH 64

union FragU { v16h v; unsigned int u[8]; };

// A-fragment (16x32 f16, MxK): lane 0-15 row M=lane, lane 16-31 row M=lane-16.
// VGPR j<4 holds K pair {2j,2j+1}+8*half; j>=4 holds {16+2(j-4),...}+8*half.
// Source: row-major f16 [row][stride] -> K pairs are contiguous, 32-bit loads.
__device__ inline v16h load_a_frag(const _Float16* base, int row0, int stride,
                                   int k0, int lane) {
  const unsigned int* p = (const unsigned int*)base;
  int half = (lane >> 4) & 1;
  int m = lane & 15;
  FragU r;
  int off = (row0 + m) * stride + k0 + 8 * half;
#pragma unroll
  for (int j = 0; j < 8; ++j) {
    int kb = (j < 4) ? (2 * j) : (16 + 2 * (j - 4));
    r.u[j] = p[(off + kb) >> 1];
  }
  return r.v;
}

// B-fragment (32x16 f16, KxN): lane holds column N=lane%16; same K order as A:
// VGPR j<4 holds K pair {2j,2j+1}+8*half; j>=4 holds {16+2(j-4),...}+8*half.
// Source is stored TRANSPOSED row-major [N][stride] so K pairs are contiguous.
__device__ inline v16h load_b_frag(const _Float16* baseT, int n0, int stride,
                                   int k0, int lane) {
  const unsigned int* p = (const unsigned int*)baseT;
  int half = (lane >> 4) & 1;
  int n = lane & 15;
  FragU r;
  int off = (n0 + n) * stride + k0 + 8 * half;
#pragma unroll
  for (int j = 0; j < 8; ++j) {
    int kb = (j < 4) ? (2 * j) : (16 + 2 * (j - 4));
    r.u[j] = p[(off + kb) >> 1];
  }
  return r.v;
}

// Convert Wq/Wk/Wv ([C][H] f32) -> transposed f16 [H][C] into workspace.
// 48 KB total; stays resident in the 192 MB L2 for all 4096 blocks.
__global__ void prep_w_kernel(const float* __restrict__ Wq,
                              const float* __restrict__ Wk,
                              const float* __restrict__ Wv,
                              _Float16* __restrict__ wt) {
  int i = blockIdx.x * blockDim.x + threadIdx.x;
  if (i >= 3 * NH * NC) return;
  int mat = i / (NH * NC);
  int r = i % (NH * NC);
  int h = r / NC, c = r % NC;
  const float* W = (mat == 0) ? Wq : ((mat == 1) ? Wk : Wv);
  wt[i] = (_Float16)W[c * NH + h];
}

__global__ __launch_bounds__(256) void attn_fused_kernel(
    const float* __restrict__ x, const _Float16* __restrict__ wt,
    float* __restrict__ out) {
  // LDS plan (56 KB, regions time-multiplexed):
  //  [0,16K)   x f16 [64][128]      -> later attn f16 [64][64]
  //  [16K,32K) W^T f16 [64][128]    -> later scores f32 [64][64]
  //  [32K,40K) q f16 [64][64] row-major
  //  [40K,48K) k f16 [64][64] row-major
  //  [48K,56K) v^T f16 [64][64] ([h][t])
  __shared__ __align__(16) char smem[57344];
  _Float16* xl  = (_Float16*)(smem);
  _Float16* wl  = (_Float16*)(smem + 16384);
  _Float16* ql  = (_Float16*)(smem + 32768);
  _Float16* kl  = (_Float16*)(smem + 40960);
  _Float16* vtl = (_Float16*)(smem + 49152);
  float*    scl = (float*)(smem + 16384);   // aliases wl (dead after proj)
  _Float16* al  = xl;                       // aliases xl (dead after proj)

  const int b    = blockIdx.x;
  const int tid  = threadIdx.x;
  const int wave = tid >> 5;
  const int lane = tid & 31;
  const int half = (lane >> 4) & 1;
  const int nlan = lane & 15;

  // ---- load x[b] (64x128 f32) -> f16 LDS, 16B global loads ----
  {
    const v4f* xv = (const v4f*)(x + (size_t)b * NT * NC);
    v4h* xo = (v4h*)xl;
    for (int i = tid; i < (NT * NC) / 4; i += 256) {
      v4f t = xv[i];
      v4h h;
      h.x = (_Float16)t.x; h.y = (_Float16)t.y;
      h.z = (_Float16)t.z; h.w = (_Float16)t.w;
      xo[i] = h;
    }
  }

  // ---- q, k, v projections: D = x(64x128) @ W(128x64) ----
  for (int pmat = 0; pmat < 3; ++pmat) {
    { // stage W^T (16 KB) into LDS from L2
      const uint4* ws = (const uint4*)(wt + pmat * (NH * NC));
      uint4* wd = (uint4*)wl;
      for (int i = tid; i < (NH * NC * 2) / 16; i += 256) wd[i] = ws[i];
    }
    __syncthreads();
    for (int i = 0; i < 2; ++i) {            // 16 tiles / 8 waves
      int idx = wave * 2 + i;
      int tt = idx >> 2, hh = idx & 3;
      v8f acc = {};
#pragma unroll
      for (int kc = 0; kc < 4; ++kc) {       // K = 128 in chunks of 32
        v16h a  = load_a_frag(xl, tt * 16, NC, kc * 32, lane);
        v16h bb = load_b_frag(wl, hh * 16, NC, kc * 32, lane);
        acc = __builtin_amdgcn_wmma_f32_16x16x32_f16(false, a, false, bb,
                                                     (short)0, acc, false, false);
      }
      if (pmat == 2) {                       // v stored transposed [h][t]
#pragma unroll
        for (int r = 0; r < 8; ++r) {
          int M = tt * 16 + r + 8 * half, N = hh * 16 + nlan;
          vtl[N * NT + M] = (_Float16)acc[r];
        }
      } else {
        _Float16* dst = (pmat == 0) ? ql : kl;
#pragma unroll
        for (int r = 0; r < 8; ++r) {
          int M = tt * 16 + r + 8 * half, N = hh * 16 + nlan;
          dst[M * NH + N] = (_Float16)acc[r];
        }
      }
    }
    __syncthreads();
  }

  // ---- scores = q @ k^T * C^-0.5  (skip strictly-upper causal tiles) ----
  const float scale = 0.08838834764831845f;  // 1/sqrt(128)
  for (int i = 0; i < 2; ++i) {
    int idx = wave * 2 + i;
    int tt = idx >> 2, ss = idx & 3;
    if (ss > tt) continue;                   // never read by softmax
    v8f acc = {};
#pragma unroll
    for (int kc = 0; kc < 2; ++kc) {         // K = H = 64
      v16h a  = load_a_frag(ql, tt * 16, NH, kc * 32, lane);
      v16h bb = load_b_frag(kl, ss * 16, NH, kc * 32, lane); // k row s = col s of k^T
      acc = __builtin_amdgcn_wmma_f32_16x16x32_f16(false, a, false, bb,
                                                   (short)0, acc, false, false);
    }
#pragma unroll
    for (int r = 0; r < 8; ++r) {
      int M = tt * 16 + r + 8 * half, N = ss * 16 + nlan;
      scl[M * NT + N] = acc[r] * scale;
    }
  }
  __syncthreads();

  // ---- causal softmax, fp32: 4 lanes per row, 16 cols each ----
  {
    int row = tid >> 2;
    int seg = tid & 3;
    float e[16];
    float m = -3.4e38f;
#pragma unroll
    for (int c = 0; c < 16; ++c) {
      int s = seg * 16 + c;
      float v = (s <= row) ? scl[row * NT + s] : -3.4e38f;
      e[c] = v;
      m = fmaxf(m, v);
    }
    m = fmaxf(m, __shfl_xor(m, 1));
    m = fmaxf(m, __shfl_xor(m, 2));
    float sum = 0.f;
#pragma unroll
    for (int c = 0; c < 16; ++c) {
      int s = seg * 16 + c;
      float ev = (s <= row) ? __expf(e[c] - m) : 0.f;
      e[c] = ev;
      sum += ev;
    }
    sum += __shfl_xor(sum, 1);
    sum += __shfl_xor(sum, 2);
    float inv = 1.0f / sum;
#pragma unroll
    for (int c = 0; c < 16; ++c) {
      int s = seg * 16 + c;
      al[row * NT + s] = (_Float16)(e[c] * inv);  // masked cols -> exact 0
    }
  }
  __syncthreads();

  // ---- out = attn(64x64) @ v(64x64), fp32 straight to HBM ----
  float* ob = out + (size_t)b * NT * NH;
  for (int i = 0; i < 2; ++i) {
    int idx = wave * 2 + i;
    int tt = idx >> 2, hh = idx & 3;
    v8f acc = {};
#pragma unroll
    for (int kc = 0; kc < 2; ++kc) {         // K = T = 64
      v16h a  = load_a_frag(al, tt * 16, NT, kc * 32, lane);
      v16h bb = load_b_frag(vtl, hh * 16, NT, kc * 32, lane); // v^T: pairs along s
      acc = __builtin_amdgcn_wmma_f32_16x16x32_f16(false, a, false, bb,
                                                   (short)0, acc, false, false);
    }
#pragma unroll
    for (int r = 0; r < 8; ++r) {
      int M = tt * 16 + r + 8 * half, N = hh * 16 + nlan;
      ob[M * NH + N] = acc[r];
    }
  }
}

extern "C" void kernel_launch(void* const* d_in, const int* in_sizes, int n_in,
                              void* d_out, int out_size, void* d_ws, size_t ws_size,
                              hipStream_t stream) {
  const float* x  = (const float*)d_in[0];
  const float* Wq = (const float*)d_in[1];
  const float* Wk = (const float*)d_in[2];
  const float* Wv = (const float*)d_in[3];
  float* out = (float*)d_out;
  _Float16* wt = (_Float16*)d_ws;  // 3 * 64 * 128 f16 = 48 KB

  prep_w_kernel<<<(3 * NH * NC + 255) / 256, 256, 0, stream>>>(Wq, Wk, Wv, wt);
  attn_fused_kernel<<<NB, 256, 0, stream>>>(x, wt, out);
}
